// TernaryLinear_6914897346865
// MI455X (gfx1250) — matrix operand order repaired and checked by reading, not compiled, not run on hardware
//
#include <hip/hip_runtime.h>

typedef __attribute__((ext_vector_type(16))) _Float16 v16h;
typedef __attribute__((ext_vector_type(8)))  _Float16 v8h;
typedef __attribute__((ext_vector_type(4)))  _Float16 v4h;
typedef __attribute__((ext_vector_type(8)))  float    v8f;
typedef __attribute__((ext_vector_type(4)))  float    v4f;

#define IN_F 256
#define OUT_F 256
#define LDS_STRIDE 264   // 256 + 8 pad halfs: row stride 132 dwords (== 4 mod 64 banks)

__global__ __launch_bounds__(256) void ternary_linear_wmma(
    const float* __restrict__ x,
    const float* __restrict__ w,
    const float* __restrict__ bias,
    float* __restrict__ out)
{
  __shared__ _Float16 wlds[OUT_F * LDS_STRIDE];   // 132 KB

  const int tid = threadIdx.x;

  // ---- Stage W (f32 -> f16, exact for {-1,0,1}) into padded LDS, coalesced ----
  #pragma unroll 4
  for (int i = tid; i < (OUT_F * IN_F) / 4; i += 256) {
    v4f wv = *(const v4f*)(w + 4 * i);
    int n = (4 * i) >> 8;       // row (out-feature)
    int k = (4 * i) & 255;      // col (in-feature)
    v4h hv;
    hv[0] = (_Float16)wv[0]; hv[1] = (_Float16)wv[1];
    hv[2] = (_Float16)wv[2]; hv[3] = (_Float16)wv[3];
    *(v4h*)(&wlds[n * LDS_STRIDE + k]) = hv;
  }
  __syncthreads();

  const int wave = tid >> 5;            // 0..7
  const int lane = tid & 31;
  const int mrow = lane & 15;           // M (and N) index within tile
  const int hsel = lane >> 4;           // lane half select

  const long row0 = (long)blockIdx.x * 128 + wave * 16;
  const float* xrow = x + (row0 + mrow) * IN_F;

  // ---- Preload all 16 per-lane bias values (keeps global loads out of hot loop)
  float bcol[16];
  #pragma unroll
  for (int t = 0; t < 16; ++t) bcol[t] = bias[16 * t + mrow];

  // ---- Load 16x256 A panel; split each element into hi/lo f16 (exact sum) ----
  v16h a_hi[8], a_lo[8];
  #pragma unroll
  for (int c = 0; c < 8; ++c) {
    const int kb = 32 * c + (hsel ? 8 : 0);
    v4f f0 = *(const v4f*)(xrow + kb);
    v4f f1 = *(const v4f*)(xrow + kb + 4);
    v4f f2 = *(const v4f*)(xrow + kb + 16);
    v4f f3 = *(const v4f*)(xrow + kb + 20);
    float fv[16] = {f0[0],f0[1],f0[2],f0[3], f1[0],f1[1],f1[2],f1[3],
                    f2[0],f2[1],f2[2],f2[3], f3[0],f3[1],f3[2],f3[3]};
    #pragma unroll
    for (int i = 0; i < 16; ++i) {
      _Float16 h = (_Float16)fv[i];
      a_hi[c][i] = h;
      a_lo[c][i] = (_Float16)(fv[i] - (float)h);
    }
  }

  float* orow = out + (row0 + (hsel ? 8 : 0)) * OUT_F;

  // ---- Sweep 16 N-tiles; A panel stays resident in VGPRs ----
  #pragma unroll 1
  for (int t = 0; t < 16; ++t) {
    const int n0 = t * 16;
    const float bval = bcol[t];           // C = bias broadcast over rows
    v8f acc;
    #pragma unroll
    for (int r = 0; r < 8; ++r) acc[r] = bval;

    #pragma unroll
    for (int c = 0; c < 8; ++c) {
      const int k0 = 32 * c + (hsel ? 8 : 0);
      const _Float16* bp = &wlds[(n0 + mrow) * LDS_STRIDE + k0];
      v8h b0 = *(const v8h*)(bp);         // K halfs k0..k0+7 and k0+16..k0+23 = B fragment
      v8h b1 = *(const v8h*)(bp + 16);
      v16h bm;
      #pragma unroll
      for (int i = 0; i < 8; ++i) { bm[i] = b0[i]; bm[i + 8] = b1[i]; }
      acc = __builtin_amdgcn_wmma_f32_16x16x32_f16(false, a_hi[c], false, bm,
                                                   (short)0, acc, false, false);
      acc = __builtin_amdgcn_wmma_f32_16x16x32_f16(false, a_lo[c], false, bm,
                                                   (short)0, acc, false, false);
    }

    // fmod(v, 2.0) truncated remainder, then coalesced row-segment stores
    #pragma unroll
    for (int r = 0; r < 8; ++r) {
      float v = acc[r];
      float q = __builtin_truncf(v * 0.5f);
      v = __builtin_fmaf(-2.0f, q, v);
      orow[(long)r * OUT_F + n0 + mrow] = v;
    }
  }
}

extern "C" void kernel_launch(void* const* d_in, const int* in_sizes, int n_in,
                              void* d_out, int out_size, void* d_ws, size_t ws_size,
                              hipStream_t stream) {
  const float* x    = (const float*)d_in[0];
  const float* wgt  = (const float*)d_in[1];
  const float* bias = (const float*)d_in[2];
  float* out = (float*)d_out;

  const int batch = in_sizes[0] / IN_F;     // 524288
  const int blocks = batch / 128;           // 4096 blocks x 128 rows
  ternary_linear_wmma<<<blocks, 256, 0, stream>>>(x, wgt, bias, out);
}
